// LongformerSelfAttention_9629316678004
// MI455X (gfx1250) — matrix operand order repaired and checked by reading, not compiled, not run on hardware
//
#include <hip/hip_runtime.h>
#include <hip/hip_bf16.h>

typedef __attribute__((ext_vector_type(16))) _Float16 v16h;
typedef __attribute__((ext_vector_type(8)))  _Float16 v8h;
typedef __attribute__((ext_vector_type(4)))  _Float16 v4h;
typedef __attribute__((ext_vector_type(8)))  float    v8f;

#define BB   2
#define SS   4096
#define EE   768
#define HH   12
#define HDIM 64
#define WW   256
#define GG   64
#define NEGM (-10000.0f)
#define NINF (-1.0e30f)

__device__ __forceinline__ int clampi(int x, int lo, int hi) {
  return x < lo ? lo : (x > hi ? hi : x);
}

// D = A(16x32 f16) * B(32x16 f16) + C(16x16 f32)
__device__ __forceinline__ v8f wmma16(v16h a, v16h b, v8f c) {
  return __builtin_amdgcn_wmma_f32_16x16x32_f16(
      /*neg_a=*/false, a, /*neg_b=*/false, b,
      /*c_mod=*/(short)0, c, /*reuse_a=*/false, /*reuse_b=*/false);
}

// ---------------- CDNA5 async global->LDS copy ----------------
// One b128 per lane: LDS byte address in a VGPR, 64-bit global address in a
// VGPR pair.  Tracked by ASYNCcnt.
__device__ __forceinline__ void async_copy_b128(unsigned lds_addr, const void* gaddr) {
  asm volatile("global_load_async_to_lds_b128 %0, %1, off"
               :
               : "v"(lds_addr), "v"(gaddr)
               : "memory");
}

__device__ __forceinline__ void wait_async0() {
#if defined(__has_builtin)
#  if __has_builtin(__builtin_amdgcn_s_wait_asynccnt)
  __builtin_amdgcn_s_wait_asynccnt(0);
#  else
  asm volatile("s_wait_asynccnt 0x0" ::: "memory");
#  endif
#else
  asm volatile("s_wait_asynccnt 0x0" ::: "memory");
#endif
}

// A fragment: caller passes this lane's row pointer (row m = lane&15).
// Contiguous layout: lane<16 holds K [0..15]; lane>=16 holds
// K [16..31].  kb = (lane<16)?0:8.
__device__ __forceinline__ v16h load_a(const _Float16* row, int kb) {
  v8h c0 = *(const v8h*)(row + 2 * kb);
  v8h c1 = *(const v8h*)(row + 2 * kb + 8);
  v16h r;
#pragma unroll
  for (int i = 0; i < 8; ++i) { r[i] = c0[i]; r[8 + i] = c1[i]; }
  return r;
}

// B fragment (32x16) sourced from a row-major [N][K] slice: B[k][n] = p[n*ld + k].
// Lanes 0-15 hold K=0..15 of column n; lanes 16-31 hold K=16..31.
__device__ __forceinline__ v16h load_bT(const _Float16* p, int ld, int lane) {
  int n  = lane & 15;
  int ko = (lane < 16) ? 0 : 16;
  return *(const v16h*)(p + (size_t)n * ld + ko);
}

// Stage one 32-key pair into LDS (all 128 threads cooperate):
//   K pair : 32 rows x 64 halfs  (4KB), row-major
//   V^T    : 64 rows x 32 halfs  (4KB), row-major (rows = head dims)
// Addresses are clamped to legal memory; garbage rows are masked to p==0 later.
__device__ __forceinline__ void stage_pair(
    _Float16* kb_lds, _Float16* vb_lds,
    const _Float16* __restrict__ kh,   // [S, HDIM]
    const _Float16* __restrict__ vh,   // [HDIM, S]
    int kt0)
{
  const int tid = threadIdx.x;  // 0..127
#pragma unroll
  for (int r = 0; r < 2; ++r) {
    int u = r * 128 + tid;            // 256 units of 16B
    int row = u >> 3, chunk = u & 7;  // row = key, 8 chunks per 64-half row
    const _Float16* g = kh + (size_t)clampi(kt0 + row, 0, SS - 1) * HDIM + chunk * 8;
    async_copy_b128((unsigned)(uintptr_t)(kb_lds + u * 8), g);
  }
#pragma unroll
  for (int r = 0; r < 2; ++r) {
    int u = r * 128 + tid;            // 256 units of 16B
    int d = u >> 2, kc = u & 3;       // row = head dim, 4 chunks per 32-half row
    const _Float16* g = vh + (size_t)d * SS + clampi(kt0 + kc * 8, 0, SS - 8);
    async_copy_b128((unsigned)(uintptr_t)(vb_lds + u * 8), g);
  }
}

// One 32-key flash step from LDS: scores (S^T via 4 WMMA), online softmax,
// PV (4 WMMA).  q0w = this wave's query-tile base.
__device__ __forceinline__ void attn_pair_lds(
    const _Float16* __restrict__ kb_lds,  // [32][HDIM]
    const _Float16* __restrict__ vb_lds,  // [HDIM][32]
    const float*    __restrict__ fm,      // [S] float mask (band only)
    v16h qf0, v16h qf1,
    int kt0, int q0w, bool band,
    float& m_i, float& l_i, v8f acc[4])
{
  const int lane = threadIdx.x & 31;
  const int kb   = (lane < 16) ? 0 : 8;
  float sc[2][8];
#pragma unroll
  for (int t = 0; t < 2; ++t) {
    int  kt  = kt0 + t * 16;
    bool oob = band && (kt < 0 || kt > SS - 16);
    const _Float16* krow = kb_lds + (size_t)(t * 16 + (lane & 15)) * HDIM;
    v16h a0 = load_a(krow, kb);        // hd 0..31  (A rows = keys)
    v16h a1 = load_a(krow + 32, kb);   // hd 32..63
    v8f s = {};
    s = wmma16(a0, qf0, s);
    s = wmma16(a1, qf1, s);            // D: rows=keys, cols=queries (S^T)
    int qcol   = q0w + (lane & 15);
    int krbase = kt + kb;
#pragma unroll
    for (int j = 0; j < 8; ++j) {
      float sv = s[j];
      if (band) {
        int  key   = krbase + j;
        int  dd    = key - qcol;
        bool valid = (!oob) && (dd >= -WW) && (dd <= WW);
        float mv   = fm[valid ? key : 0];
        sv = valid ? (sv + mv) : NINF;
      }
      sc[t][j] = sv;
    }
  }
  // per-query max over the 32 keys: 16 in-lane values + partner half
  float tmax = NINF;
#pragma unroll
  for (int j = 0; j < 8; ++j) tmax = fmaxf(tmax, fmaxf(sc[0][j], sc[1][j]));
  tmax = fmaxf(tmax, __shfl_xor(tmax, 16, 32));
  float m_new = fmaxf(m_i, tmax);
  float alpha = __expf(m_i - m_new);
  float p0[8], p1[8];
  float rsum = 0.0f;
#pragma unroll
  for (int j = 0; j < 8; ++j) {
    p0[j] = __expf(sc[0][j] - m_new);
    p1[j] = __expf(sc[1][j] - m_new);
    rsum += p0[j] + p1[j];
  }
  rsum += __shfl_xor(rsum, 16, 32);
  l_i = l_i * alpha + rsum;
  m_i = m_new;
#pragma unroll
  for (int dt = 0; dt < 4; ++dt)
#pragma unroll
    for (int j = 0; j < 8; ++j) acc[dt][j] *= alpha;
  // Build P^T B-fragment (K=32 keys, N=16 queries) from the two S^T tiles.
  v16h pf;
#pragma unroll
  for (int j = 0; j < 8; ++j) {
    float o0 = __shfl_xor(p0[j], 16, 32);
    float o1 = __shfl_xor(p1[j], 16, 32);
    pf[j]     = (_Float16)((lane < 16) ? p0[j] : o1);
    pf[8 + j] = (_Float16)((lane < 16) ? o0 : p1[j]);
  }
  // O^T += V^T * P^T  (4 d-tiles of 16 dims each); V^T rows are 32 halfs wide.
#pragma unroll
  for (int dt = 0; dt < 4; ++dt) {
    const _Float16* vrow = vb_lds + (size_t)(dt * 16 + (lane & 15)) * 32;
    v16h af = load_a(vrow, kb);
    acc[dt] = wmma16(af, pf, acc[dt]);
  }
}

// ---------------- elementwise prep kernels ----------------

__global__ void cvt_kernel(const float* __restrict__ s, _Float16* __restrict__ d, long n4) {
  long i = (long)blockIdx.x * blockDim.x + threadIdx.x;
  if (i >= n4) return;
  float4 x = ((const float4*)s)[i];
  v4h y;
  y[0] = (_Float16)x.x; y[1] = (_Float16)x.y;
  y[2] = (_Float16)x.z; y[3] = (_Float16)x.w;
  ((v4h*)d)[i] = y;
}

__global__ void mask_kernel(const int* __restrict__ m, float* __restrict__ fm) {
  int i = blockIdx.x * blockDim.x + threadIdx.x;
  if (i < BB * SS) fm[i] = (m[i] != 0) ? NEGM : 0.0f;
}

// [B,H,S,HD] -> [B,H,HD,S]
__global__ void vtrans_kernel(const _Float16* __restrict__ src, _Float16* __restrict__ dst) {
  size_t i = (size_t)blockIdx.x * blockDim.x + threadIdx.x;
  if (i >= (size_t)BB * HH * SS * HDIM) return;
  int    d  = (int)(i & (HDIM - 1));
  size_t r  = i >> 6;
  int    s  = (int)(r & (SS - 1));
  size_t bh = r >> 12;
  dst[(bh * HDIM + d) * SS + s] = src[i];
}

// ---------------- projection GEMM (WMMA) ----------------
// o[b, e/64, s, e%64] = (sum_c x[b,s,c] * W[e,c] + bias[e]) * scale
__global__ void __launch_bounds__(128) proj_kernel(
    const _Float16* __restrict__ xh,   // [B,S,E]
    const _Float16* __restrict__ wh,   // [E,E] row-major (row = out feature)
    const float*    __restrict__ bias, // [E]
    _Float16*       __restrict__ o,    // [B,H,rows,HDIM]
    int rows, float scale)
{
  const int lane = threadIdx.x & 31;
  const int wave = threadIdx.x >> 5;
  const int s0   = blockIdx.x * 16;
  const int e0   = blockIdx.y * 64 + wave * 16;
  const int b    = blockIdx.z;
  const int kb   = (lane < 16) ? 0 : 8;
  const _Float16* xrow = xh + ((size_t)b * SS + (s0 + (lane & 15))) * EE;
  v8f acc = {};
  for (int kc = 0; kc < EE; kc += 32) {
    v16h a  = load_a(xrow + kc, kb);
    v16h bf = load_bT(wh + (size_t)e0 * EE + kc, EE, lane);
    acc = wmma16(a, bf, acc);
  }
  const int   n  = lane & 15;
  const int   e  = e0 + n;
  const int   hh = e >> 6;
  const int   d  = e & 63;
  const float bv = bias[e];
  const int   m0 = (lane < 16) ? 0 : 8;
  _Float16* dst = o + (((size_t)b * HH + hh) * rows + s0) * HDIM + d;
#pragma unroll
  for (int j = 0; j < 8; ++j)
    dst[(size_t)(m0 + j) * HDIM] = (_Float16)((acc[j] + bv) * scale);
}

// ---------------- band + global-key attention ----------------
// Block = 64 queries (4 waves x 16).  All waves share each staged 32-key pair.
__global__ void __launch_bounds__(128) band_attn_kernel(
    const _Float16* __restrict__ q,     // [B,H,S,HD], pre-scaled
    const _Float16* __restrict__ k,     // [B,H,S,HD]
    const _Float16* __restrict__ vt,    // [B,H,HD,S]
    const float*    __restrict__ fmask, // [B,S]
    float*          __restrict__ out)   // [B,S,E]
{
  __shared__ _Float16 kbuf[2][32 * HDIM];
  __shared__ _Float16 vbuf[2][HDIM * 32];

  const int lane = threadIdx.x & 31;
  const int wave = threadIdx.x >> 5;
  const int Q0   = blockIdx.x * 64;      // block query base
  const int h    = blockIdx.y;
  const int b    = blockIdx.z;
  const int qw   = Q0 + wave * 16;       // this wave's query tile

  const _Float16* qh = q  + ((size_t)(b * HH + h) * SS) * HDIM;
  const _Float16* kh = k  + ((size_t)(b * HH + h) * SS) * HDIM;
  const _Float16* vh = vt + ((size_t)(b * HH + h) * HDIM) * SS;
  const float*    fm = fmask + (size_t)b * SS;

  v16h qf0 = load_bT(qh + (size_t)qw * HDIM,      HDIM, lane);  // hd 0..31
  v16h qf1 = load_bT(qh + (size_t)qw * HDIM + 32, HDIM, lane);  // hd 32..63

  float m_i = NINF, l_i = 0.0f;
  v8f acc[4] = { {}, {}, {}, {} };

  // pairs 0..1: global keys [0,64); pairs 2..19: band union [Q0-256, Q0+320)
  const int NP = 20;
  auto pair_kt0 = [&](int i) -> int { return (i < 2) ? 32 * i : (Q0 - WW + 32 * (i - 2)); };

  stage_pair(kbuf[0], vbuf[0], kh, vh, pair_kt0(0));
#pragma unroll 1
  for (int i = 0; i < NP; ++i) {
    const int cur = i & 1;
    wait_async0();       // my async writes into kbuf[cur]/vbuf[cur] landed
    __syncthreads();     // everyone's writes landed; prior reads of buffers done
    if (i + 1 < NP)
      stage_pair(kbuf[cur ^ 1], vbuf[cur ^ 1], kh, vh, pair_kt0(i + 1));
    attn_pair_lds(kbuf[cur], vbuf[cur], fm, qf0, qf1,
                  pair_kt0(i), qw, i >= 2, m_i, l_i, acc);
  }

  const float inv = 1.0f / l_i;
  const int n  = lane & 15;
  const int m0 = (lane < 16) ? 0 : 8;
  float* orow = out + ((size_t)b * SS + (qw + n)) * EE + h * HDIM;
#pragma unroll
  for (int dt = 0; dt < 4; ++dt) {
    float4 lo, hi;
    lo.x = acc[dt][0] * inv; lo.y = acc[dt][1] * inv;
    lo.z = acc[dt][2] * inv; lo.w = acc[dt][3] * inv;
    hi.x = acc[dt][4] * inv; hi.y = acc[dt][5] * inv;
    hi.z = acc[dt][6] * inv; hi.w = acc[dt][7] * inv;
    *(float4*)(orow + dt * 16 + m0)     = lo;
    *(float4*)(orow + dt * 16 + m0 + 4) = hi;
  }
}

// ---------------- global-token attention (rows [0,G)) ----------------
__global__ void __launch_bounds__(128) global_attn_kernel(
    const _Float16* __restrict__ qg,   // [B,H,G,HD], pre-scaled
    const _Float16* __restrict__ kg,   // [B,H,S,HD]
    const _Float16* __restrict__ vgt,  // [B,H,HD,S]
    const float*    __restrict__ fmask,
    float*          __restrict__ out)
{
  __shared__ _Float16 kbuf[2][32 * HDIM];
  __shared__ _Float16 vbuf[2][HDIM * 32];

  const int lane = threadIdx.x & 31;
  const int wave = threadIdx.x >> 5;
  const int qw   = wave * 16;          // 4 waves -> G=64 queries
  const int h    = blockIdx.y;
  const int b    = blockIdx.z;

  const _Float16* qh = qg  + ((size_t)(b * HH + h) * GG) * HDIM;
  const _Float16* kh = kg  + ((size_t)(b * HH + h) * SS) * HDIM;
  const _Float16* vh = vgt + ((size_t)(b * HH + h) * HDIM) * SS;

  v16h qf0 = load_bT(qh + (size_t)qw * HDIM,      HDIM, lane);
  v16h qf1 = load_bT(qh + (size_t)qw * HDIM + 32, HDIM, lane);

  float m_i = NINF, l_i = 0.0f;
  v8f acc[4] = { {}, {}, {}, {} };

  const int NP = SS / 32;  // 128 pairs over all keys
  stage_pair(kbuf[0], vbuf[0], kh, vh, 0);
#pragma unroll 1
  for (int i = 0; i < NP; ++i) {
    const int cur = i & 1;
    wait_async0();
    __syncthreads();
    if (i + 1 < NP)
      stage_pair(kbuf[cur ^ 1], vbuf[cur ^ 1], kh, vh, 32 * (i + 1));
    attn_pair_lds(kbuf[cur], vbuf[cur], fmask, qf0, qf1,
                  32 * i, qw, false, m_i, l_i, acc);
  }

  const float inv = 1.0f / l_i;
  const int n  = lane & 15;
  const int m0 = (lane < 16) ? 0 : 8;
  float* orow = out + ((size_t)b * SS + (qw + n)) * EE + h * HDIM;
#pragma unroll
  for (int dt = 0; dt < 4; ++dt) {
    float4 lo, hi;
    lo.x = acc[dt][0] * inv; lo.y = acc[dt][1] * inv;
    lo.z = acc[dt][2] * inv; lo.w = acc[dt][3] * inv;
    hi.x = acc[dt][4] * inv; hi.y = acc[dt][5] * inv;
    hi.z = acc[dt][6] * inv; hi.w = acc[dt][7] * inv;
    *(float4*)(orow + dt * 16 + m0)     = lo;
    *(float4*)(orow + dt * 16 + m0 + 4) = hi;
  }
}

// ---------------- host launcher ----------------
extern "C" void kernel_launch(void* const* d_in, const int* in_sizes, int n_in,
                              void* d_out, int out_size, void* d_ws, size_t ws_size,
                              hipStream_t stream)
{
  (void)in_sizes; (void)n_in; (void)out_size; (void)ws_size;
  const float* x   = (const float*)d_in[0];
  const float* Wq  = (const float*)d_in[1];  const float* bq  = (const float*)d_in[2];
  const float* Wk  = (const float*)d_in[3];  const float* bk  = (const float*)d_in[4];
  const float* Wv  = (const float*)d_in[5];  const float* bv  = (const float*)d_in[6];
  const float* Wqg = (const float*)d_in[7];  const float* bqg = (const float*)d_in[8];
  const float* Wkg = (const float*)d_in[9];  const float* bkg = (const float*)d_in[10];
  const float* Wvg = (const float*)d_in[11]; const float* bvg = (const float*)d_in[12];
  const int*   am  = (const int*)d_in[13];
  float* out = (float*)d_out;

  char* ws  = (char*)d_ws;
  size_t off = 0;
  auto take = [&](size_t bytes) -> char* {
    char* p = ws + off;
    off += (bytes + 255) & ~(size_t)255;
    return p;
  };
  const size_t NX = (size_t)BB * SS * EE;   // 6,291,456 elements
  const size_t NW = (size_t)EE * EE;        // 589,824

  _Float16* XH   = (_Float16*)take(NX * 2);
  _Float16* WHq  = (_Float16*)take(NW * 2);
  _Float16* WHk  = (_Float16*)take(NW * 2);
  _Float16* WHv  = (_Float16*)take(NW * 2);
  _Float16* WHqg = (_Float16*)take(NW * 2);
  _Float16* WHkg = (_Float16*)take(NW * 2);
  _Float16* WHvg = (_Float16*)take(NW * 2);
  _Float16* Qb   = (_Float16*)take(NX * 2);
  _Float16* Kb   = (_Float16*)take(NX * 2);
  _Float16* Vb   = (_Float16*)take(NX * 2);
  _Float16* KGb  = (_Float16*)take(NX * 2);
  _Float16* VGb  = (_Float16*)take(NX * 2);
  _Float16* QGb  = (_Float16*)take((size_t)BB * HH * GG * HDIM * 2);
  _Float16* VT   = (_Float16*)take(NX * 2);
  _Float16* VGT  = (_Float16*)take(NX * 2);
  float*    FM   = (float*)take((size_t)BB * SS * 4);

  const long n4x = (long)(NX / 4), n4w = (long)(NW / 4);
  cvt_kernel<<<dim3((unsigned)((n4x + 255) / 256)), 256, 0, stream>>>(x,   XH,   n4x);
  cvt_kernel<<<dim3((unsigned)((n4w + 255) / 256)), 256, 0, stream>>>(Wq,  WHq,  n4w);
  cvt_kernel<<<dim3((unsigned)((n4w + 255) / 256)), 256, 0, stream>>>(Wk,  WHk,  n4w);
  cvt_kernel<<<dim3((unsigned)((n4w + 255) / 256)), 256, 0, stream>>>(Wv,  WHv,  n4w);
  cvt_kernel<<<dim3((unsigned)((n4w + 255) / 256)), 256, 0, stream>>>(Wqg, WHqg, n4w);
  cvt_kernel<<<dim3((unsigned)((n4w + 255) / 256)), 256, 0, stream>>>(Wkg, WHkg, n4w);
  cvt_kernel<<<dim3((unsigned)((n4w + 255) / 256)), 256, 0, stream>>>(Wvg, WHvg, n4w);
  mask_kernel<<<dim3((BB * SS + 255) / 256), 256, 0, stream>>>(am, FM);

  const float qscale = 0.125f;  // 1/sqrt(HD)
  dim3 gp(SS / 16, EE / 64, BB);
  proj_kernel<<<gp, 128, 0, stream>>>(XH, WHq,  bq,  Qb,  SS, qscale);
  proj_kernel<<<gp, 128, 0, stream>>>(XH, WHk,  bk,  Kb,  SS, 1.0f);
  proj_kernel<<<gp, 128, 0, stream>>>(XH, WHv,  bv,  Vb,  SS, 1.0f);
  proj_kernel<<<gp, 128, 0, stream>>>(XH, WHkg, bkg, KGb, SS, 1.0f);
  proj_kernel<<<gp, 128, 0, stream>>>(XH, WHvg, bvg, VGb, SS, 1.0f);
  dim3 gq(GG / 16, EE / 64, BB);
  proj_kernel<<<gq, 128, 0, stream>>>(XH, WHqg, bqg, QGb, GG, qscale);

  vtrans_kernel<<<dim3((unsigned)((NX + 255) / 256)), 256, 0, stream>>>(Vb,  VT);
  vtrans_kernel<<<dim3((unsigned)((NX + 255) / 256)), 256, 0, stream>>>(VGb, VGT);

  dim3 gb(SS / 64, HH, BB);
  band_attn_kernel<<<gb, 128, 0, stream>>>(Qb, Kb, VT, FM, out);
  dim3 gg(1, HH, BB);
  global_attn_kernel<<<gg, 128, 0, stream>>>(QGb, KGb, VGT, FM, out);
}
